// MultiAttentionhead_62380105008292
// MI455X (gfx1250) — compile-verified, matrix operand order checked by reading, not run on hardware
//
#include <hip/hip_runtime.h>
#include <hip/hip_bf16.h>

// CDNA5 (gfx1250) wave32 WMMA types
typedef __attribute__((ext_vector_type(16))) __bf16 v16bf;
typedef __attribute__((ext_vector_type(8)))  float  v8f;

#define EMBED 768
#define HEADS 8
#define HDIM  96
#define SEQ   1024
#define BATCH 16
#define ROWS  (BATCH * SEQ)          // 16384

#if __has_builtin(__builtin_amdgcn_global_load_async_to_lds_b128)
#define HAS_ASYNC_LDS 1
#else
#define HAS_ASYNC_LDS 0
#endif

// Pointer types matching the async-to-LDS builtin's signature:
// (AS1 int4*, AS3 int4*, imm offset, imm cpol)
typedef int i4v __attribute__((vector_size(16)));
typedef __attribute__((address_space(1))) i4v* g_i4v_ptr;
typedef __attribute__((address_space(3))) i4v* l_i4v_ptr;

__device__ __forceinline__ unsigned pack_bf16(float a, float b) {
    union { __bf16 h[2]; unsigned u; } r;
    r.h[0] = (__bf16)a;
    r.h[1] = (__bf16)b;
    return r.u;
}

__device__ __forceinline__ unsigned short bf16_bits(float f) {
    union { __bf16 h; unsigned short u; } r;
    r.h = (__bf16)f;
    return r.u;
}

// 16-byte global -> LDS copy. Uses the gfx1250 async-to-LDS path (ASYNCcnt)
// when the builtin exists; otherwise a plain b128 load/ds_store pair.
__device__ __forceinline__ void copy16(const void* g, void* l) {
#if HAS_ASYNC_LDS
    __builtin_amdgcn_global_load_async_to_lds_b128(
        (g_i4v_ptr)(size_t)g, (l_i4v_ptr)l, 0, 0);
#else
    *(uint4*)l = *(const uint4*)g;
#endif
}

__device__ __forceinline__ void async_join() {
#if HAS_ASYNC_LDS
#if __has_builtin(__builtin_amdgcn_s_wait_asynccnt)
    __builtin_amdgcn_s_wait_asynccnt(0);
#else
    asm volatile("s_wait_asynccnt 0" ::: "memory");
#endif
#endif
}

// Load one 16x32 bf16 fragment (ISA 16-bit A-matrix layout) from LDS stored
// as packed uint pairs, row-major [row][kdw], row stride `stride` dwords
// (stride must be ≡ 0 mod 4 so each 4-dword group is 16B aligned -> b128).
// `row` must already include (lane & 15). kdw_base = k_offset/2.
__device__ __forceinline__ v16bf load_frag(const unsigned* lds, int row,
                                           int stride, int kdw_base) {
    union { unsigned u[8]; v16bf v; } r;
    const int lane  = threadIdx.x & 31;
    const int half4 = (lane >> 4) << 2;   // 0 or 4
    const unsigned* p = lds + row * stride + kdw_base + half4;
    r.u[0] = p[0];  r.u[1] = p[1];  r.u[2] = p[2];  r.u[3] = p[3];
    r.u[4] = p[8];  r.u[5] = p[9];  r.u[6] = p[10]; r.u[7] = p[11];
    return r.v;
}

__device__ __forceinline__ v8f wmma_bf16(v16bf a, v16bf b, v8f c) {
    return __builtin_amdgcn_wmma_f32_16x16x32_bf16(
        false, a, false, b, (short)0, c, false, false);
}

// ---------------------------------------------------------------------------
// 128x128-tile GEMM, K-chunks of 32.
// MODE 0: A = x (fp32), scatter bf16 into Q [bh][tok][d] and Kt/Vt [bh][d][tok]
// MODE 1: A = Z (bf16, async-staged), plain fp32 store to Cf (N = 768)
// 8 waves as 2(M) x 4(N); each wave 64x32 = 4x2 WMMA tiles.
// ---------------------------------------------------------------------------
template <int MODE>
__global__ __launch_bounds__(256)
void gemm_wmma(const void* __restrict__ Ap, const float* __restrict__ B,
               const float* __restrict__ bias, float* __restrict__ Cf,
               unsigned short* __restrict__ Qb, unsigned short* __restrict__ Ktb,
               unsigned short* __restrict__ Vtb, int K, int N) {
    __shared__ unsigned lsA[128 * 20];   // [m][kdw], stride 20 (16B aligned, no bank conflicts)
    __shared__ unsigned lsB[128 * 20];   // [n][kdw] (B transposed)

    const int m0    = blockIdx.x * 128;
    const int n0    = blockIdx.y * 128;
    const int t     = threadIdx.x;
    const int lane  = t & 31;
    const int wave  = t >> 5;
    const int waveM = wave >> 2;     // 0..1
    const int waveN = wave & 3;      // 0..3

    const v8f vzero = {0.f, 0.f, 0.f, 0.f, 0.f, 0.f, 0.f, 0.f};
    v8f acc[4][2];
    for (int mt = 0; mt < 4; ++mt)
        for (int nt = 0; nt < 2; ++nt) acc[mt][nt] = vzero;

    for (int k0 = 0; k0 < K; k0 += 32) {
        if constexpr (MODE == 0) {
            // A tile: fp32 -> bf16, float4 (b128) global loads
            const float* Af = (const float*)Ap;
            for (int p = t; p < 1024; p += 256) {
                const int row = p >> 3, q = p & 7;
                const float4 g = *(const float4*)(Af + (size_t)(m0 + row) * K + k0 + q * 4);
                lsA[row * 20 + q * 2]     = pack_bf16(g.x, g.y);
                lsA[row * 20 + q * 2 + 1] = pack_bf16(g.z, g.w);
            }
        } else {
            // A tile is already bf16 and row-contiguous: async copy to LDS
            const char* Az = (const char*)Ap;
            for (int p = t; p < 512; p += 256) {
                const int row = p >> 2, c = p & 3;
                copy16(Az + (size_t)(m0 + row) * (K * 2) + k0 * 2 + c * 16,
                       lsA + row * 20 + c * 4);
            }
        }
        // B tile transposed [n][kdw]: fp32 -> bf16, float4 loads along n
        for (int p = t; p < 512; p += 256) {
            const int n4 = p & 31, dw = p >> 5;
            const int kk = k0 + dw * 2;
            const float4 g0 = *(const float4*)(B + (size_t)kk * N + n0 + n4 * 4);
            const float4 g1 = *(const float4*)(B + (size_t)(kk + 1) * N + n0 + n4 * 4);
            lsB[(n4 * 4 + 0) * 20 + dw] = pack_bf16(g0.x, g1.x);
            lsB[(n4 * 4 + 1) * 20 + dw] = pack_bf16(g0.y, g1.y);
            lsB[(n4 * 4 + 2) * 20 + dw] = pack_bf16(g0.z, g1.z);
            lsB[(n4 * 4 + 3) * 20 + dw] = pack_bf16(g0.w, g1.w);
        }
        if constexpr (MODE == 1) async_join();
        __syncthreads();

        if constexpr (MODE == 0) {
            if (k0 + 32 < K) {   // prefetch next K-tile (global_prefetch_b8)
                __builtin_prefetch((const float*)Ap + (size_t)(m0 + (t >> 1)) * K + k0 + 32, 0, 1);
                __builtin_prefetch(B + (size_t)(k0 + 32 + (t & 31)) * N + n0 + (t >> 5) * 16, 0, 1);
            }
        }

        v16bf fa[4], fb[2];
        for (int mt = 0; mt < 4; ++mt)
            fa[mt] = load_frag(lsA, waveM * 64 + mt * 16 + (lane & 15), 20, 0);
        for (int nt = 0; nt < 2; ++nt)
            fb[nt] = load_frag(lsB, waveN * 32 + nt * 16 + (lane & 15), 20, 0);

        for (int mt = 0; mt < 4; ++mt)
            for (int nt = 0; nt < 2; ++nt)
                acc[mt][nt] = wmma_bf16(fa[mt], fb[nt], acc[mt][nt]);
        __syncthreads();
    }

    // Epilogue. C layout: VGPR i, lane L -> row = i + 8*(L>>4), col = L&15.
    const int rsub = (lane >> 4) << 3;
    const int csub = lane & 15;
    for (int mt = 0; mt < 4; ++mt) {
        for (int nt = 0; nt < 2; ++nt) {
            const int cbase = n0 + waveN * 32 + nt * 16 + csub;
            const float bv = bias[cbase];
            for (int i = 0; i < 8; ++i) {
                const int r = m0 + waveM * 64 + mt * 16 + rsub + i;
                const float v = acc[mt][nt][i] + bv;
                if constexpr (MODE == 0) {
                    // qkv column c = h*288 + dd*3 + s
                    const int bb  = r >> 10, n = r & 1023;
                    const int h   = cbase / 288;
                    const int rem = cbase - h * 288;
                    const int dd  = rem / 3;
                    const int s   = rem - dd * 3;
                    const int bh  = bb * HEADS + h;
                    const unsigned short hv = bf16_bits(v);
                    if (s == 0)      Qb [((size_t)bh * SEQ + n) * HDIM + dd] = hv;
                    else if (s == 1) Ktb[((size_t)bh * HDIM + dd) * SEQ + n] = hv;
                    else             Vtb[((size_t)bh * HDIM + dd) * SEQ + n] = hv;
                } else {
                    Cf[(size_t)r * N + cbase] = v;
                }
            }
        }
    }
}

// ---------------------------------------------------------------------------
// KVt[bh] = transpose(K^T V): one block per (b,h). K/V stored [bh][d][tok]
// (bf16), so each 32-token chunk is a contiguous async copy into LDS.
// Waves 0..5 each own a 16-row M-strip with 6 N-tiles.
// ---------------------------------------------------------------------------
__global__ __launch_bounds__(256)
void ktv_kernel(const unsigned short* __restrict__ Kt,
                const unsigned short* __restrict__ Vt,
                unsigned short* __restrict__ KVt) {
    const int bh = blockIdx.x;
    const char* Kb = (const char*)(Kt + (size_t)bh * HDIM * SEQ);
    const char* Vb = (const char*)(Vt + (size_t)bh * HDIM * SEQ);

    __shared__ unsigned lsK[96 * 20];   // [d][tok_dw]
    __shared__ unsigned lsV[96 * 20];

    const int t = threadIdx.x, lane = t & 31, wave = t >> 5;

    const v8f vzero = {0.f, 0.f, 0.f, 0.f, 0.f, 0.f, 0.f, 0.f};
    v8f acc[6];
    for (int nt = 0; nt < 6; ++nt) acc[nt] = vzero;

    for (int k0 = 0; k0 < SEQ; k0 += 32) {
        for (int p = t; p < 768; p += 256) {       // 2 mats x 96 rows x 4 chunks
            const int mat = p / 384;
            const int q   = p - mat * 384;
            const int m   = q >> 2, c = q & 3;
            const char* src = (mat ? Vb : Kb) + (size_t)m * (SEQ * 2) + k0 * 2 + c * 16;
            unsigned*   dst = (mat ? lsV : lsK) + m * 20 + c * 4;
            copy16(src, dst);
        }
        async_join();
        __syncthreads();
        if (wave < 6) {
            const v16bf a = load_frag(lsK, wave * 16 + (lane & 15), 20, 0);
            for (int nt = 0; nt < 6; ++nt) {
                const v16bf b = load_frag(lsV, nt * 16 + (lane & 15), 20, 0);
                acc[nt] = wmma_bf16(a, b, acc[nt]);
            }
        }
        __syncthreads();
    }

    if (wave < 6) {
        const int rsub = (lane >> 4) << 3;
        const int csub = lane & 15;
        for (int nt = 0; nt < 6; ++nt)
            for (int i = 0; i < 8; ++i) {
                const int row = wave * 16 + rsub + i;   // K's d index (GEMM M)
                const int col = nt * 16 + csub;         // V's d index (GEMM N)
                // store transposed: KVt[n][k] so the next GEMM's B-stage is contiguous
                KVt[(size_t)bh * HDIM * HDIM + col * HDIM + row] = bf16_bits(acc[nt][i]);
            }
    }
}

// ---------------------------------------------------------------------------
// Z tile (128 tokens x 96) = Q_tile(128x96, bf16) * KV(96x96), K = 96.
// Both tiles are contiguous bf16 -> async copies. grid = (8 mblk, 128 bh).
// 8 waves as 4(M) x 2(N); each wave 32x48 = 2x3 tiles.
// ---------------------------------------------------------------------------
__global__ __launch_bounds__(256)
void qz_kernel(const unsigned short* __restrict__ Qbuf,
               const unsigned short* __restrict__ KVt,
               unsigned short* __restrict__ Z) {
    const int mblk = blockIdx.x;           // 0..7
    const int bh   = blockIdx.y;           // 0..127
    const int b    = bh >> 3, h = bh & 7;

    const char* Qm  = (const char*)(Qbuf + (size_t)bh * SEQ * HDIM + (size_t)mblk * 128 * HDIM);
    const char* KVm = (const char*)(KVt + (size_t)bh * HDIM * HDIM);

    __shared__ unsigned lsA[128 * 52];     // [tok][kdw], stride 52 (16B aligned, conflict-free)
    __shared__ unsigned lsB[96 * 52];      // [n][kdw]

    const int t = threadIdx.x, lane = t & 31, wave = t >> 5;
    const int waveM = wave >> 1;           // 0..3
    const int waveN = wave & 1;            // 0..1

    for (int p = t; p < 1536; p += 256) {  // 128 rows x 12 chunks of 16B
        const int row = p / 12, c = p % 12;
        copy16(Qm + (size_t)row * 192 + c * 16, (char*)lsA + row * 208 + c * 16);
    }
    for (int p = t; p < 1152; p += 256) {  // 96 rows x 12 chunks
        const int n = p / 12, c = p % 12;
        copy16(KVm + (size_t)n * 192 + c * 16, (char*)lsB + n * 208 + c * 16);
    }
    async_join();
    __syncthreads();

    const v8f vzero = {0.f, 0.f, 0.f, 0.f, 0.f, 0.f, 0.f, 0.f};
    v8f acc[2][3];
    for (int mt = 0; mt < 2; ++mt)
        for (int nt = 0; nt < 3; ++nt) acc[mt][nt] = vzero;

    for (int kc = 0; kc < 3; ++kc) {       // K window = kc*32 .. +32
        v16bf fa[2], fb[3];
        for (int mt = 0; mt < 2; ++mt)
            fa[mt] = load_frag(lsA, waveM * 32 + mt * 16 + (lane & 15), 52, kc * 16);
        for (int nt = 0; nt < 3; ++nt)
            fb[nt] = load_frag(lsB, waveN * 48 + nt * 16 + (lane & 15), 52, kc * 16);
        for (int mt = 0; mt < 2; ++mt)
            for (int nt = 0; nt < 3; ++nt)
                acc[mt][nt] = wmma_bf16(fa[mt], fb[nt], acc[mt][nt]);
    }

    const int rsub = (lane >> 4) << 3;
    const int csub = lane & 15;
    for (int mt = 0; mt < 2; ++mt)
        for (int nt = 0; nt < 3; ++nt)
            for (int i = 0; i < 8; ++i) {
                const int tok = mblk * 128 + waveM * 32 + mt * 16 + rsub + i;
                const int c   = waveN * 48 + nt * 16 + csub;
                Z[((size_t)b * SEQ + tok) * EMBED + h * HDIM + c] =
                    bf16_bits(acc[mt][nt][i]);
            }
}

// ---------------------------------------------------------------------------
extern "C" void kernel_launch(void* const* d_in, const int* in_sizes, int n_in,
                              void* d_out, int out_size, void* d_ws, size_t ws_size,
                              hipStream_t stream) {
    const float* x     = (const float*)d_in[0];   // (16,1024,768)
    const float* W_qkv = (const float*)d_in[1];   // (768,2304)
    const float* b_qkv = (const float*)d_in[2];   // (2304,)
    const float* W_fc  = (const float*)d_in[3];   // (768,768)
    const float* b_fc  = (const float*)d_in[4];   // (768,)
    float* out = (float*)d_out;

    unsigned short* wsu = (unsigned short*)d_ws;
    const size_t SZ = (size_t)BATCH * HEADS * SEQ * HDIM;  // 12,582,912 elems
    unsigned short* Q   = wsu;             // [bh][tok][d]
    unsigned short* Kt  = wsu + SZ;        // [bh][d][tok]
    unsigned short* Vt  = wsu + 2 * SZ;    // [bh][d][tok]
    unsigned short* KVt = wsu + 3 * SZ;    // [bh][n][k], 128*96*96
    unsigned short* Z   = Kt;              // K dead after ktv_kernel; reuse

    // 1) qkv = x @ W_qkv + b, scattered to bf16 Q / K^T / V^T
    gemm_wmma<0><<<dim3(ROWS / 128, (3 * EMBED) / 128), 256, 0, stream>>>(
        x, W_qkv, b_qkv, nullptr, Q, Kt, Vt, EMBED, 3 * EMBED);

    // 2) KVt[bh] = (K^T @ V)^T  (96x96 per head, bf16)
    ktv_kernel<<<BATCH * HEADS, 256, 0, stream>>>(Kt, Vt, KVt);

    // 3) Z = Q @ KV, laid out (b, n, h*d), bf16
    qz_kernel<<<dim3(SEQ / 128, BATCH * HEADS), 256, 0, stream>>>(Q, KVt, Z);

    // 4) out = Z @ W_fc + b_fc (fp32 out)
    gemm_wmma<1><<<dim3(ROWS / 128, EMBED / 128), 256, 0, stream>>>(
        Z, W_fc, b_fc, out, nullptr, nullptr, nullptr, EMBED, EMBED);
}
